// Gate_12489764896829
// MI455X (gfx1250) — matrix operand order repaired and checked by reading, not compiled, not run on hardware
//
#include <hip/hip_runtime.h>

// ---------------------------------------------------------------------------
// MoE gating for MI455X (gfx1250, wave32, WMMA).
//   logits = x[tokens,1024] @ w[1024,64]  (fp32 inputs)
//   top-8 per row (sorted desc, ties -> lower index), softmax over top-8.
// GEMM runs as split-bf16 (hi/lo) on v_wmma_f32_16x16x32_bf16:
//   x*w ~= x_hi*w_hi + x_lo*w_hi + x_hi*w_lo   (~fp32 accuracy)
// x is streamed from HBM exactly once (memory-bound at ~11us on 23.3 TB/s).
// ---------------------------------------------------------------------------

typedef __attribute__((ext_vector_type(16))) __bf16 v16bf;
typedef __attribute__((ext_vector_type(8)))  __bf16 v8bf;
typedef __attribute__((ext_vector_type(8)))  float  v8f;

#define D        1024      // model dim (K)
#define NE       64        // experts (N)
#define TOPK     8
#define SLABK    128       // K-slab staged in LDS
#define PITCH    (SLABK + 8)   // bf16 elems per LDS row (pad kills bank conflicts)
#define NSLAB    (D / SLABK)
#define WAVES_PER_WG 8
#define THREADS  (WAVES_PER_WG * 32)

__global__ __launch_bounds__(THREADS)
void moe_gate_wmma(const float* __restrict__ x,
                   const float* __restrict__ w,
                   float* __restrict__ outS,
                   int*   __restrict__ outE,
                   int ntiles)
{
  // Transposed, bf16-split weight slab: sHi/sLo[n][k_in_slab]
  __shared__ __attribute__((aligned(32))) __bf16 sHi[NE * PITCH];
  __shared__ __attribute__((aligned(32))) __bf16 sLo[NE * PITCH];

  const int tid    = threadIdx.x;
  const int lane   = tid & 31;
  const int waveId = tid >> 5;
  const int hl     = lane >> 4;    // half of wave (0: lanes 0-15, 1: lanes 16-31)
  const int l16    = lane & 15;

  for (int tb = blockIdx.x * WAVES_PER_WG; tb < ntiles; tb += gridDim.x * WAVES_PER_WG) {
    const int  tile   = tb + waveId;
    const bool active = (tile < ntiles);
    const int  row    = (active ? tile : 0) * 16 + l16;  // A-matrix row for this lane
    const float* aRow = x + (size_t)row * D;
    const int  koffA  = hl * 8;    // A 16-bit layout: lanes0-15 K{0..7,16..23}, lanes16-31 +8

    v8f c[4];
    #pragma unroll
    for (int t = 0; t < 4; ++t) c[t] = (v8f){0.f,0.f,0.f,0.f,0.f,0.f,0.f,0.f};

    for (int slab = 0; slab < NSLAB; ++slab) {
      const int kk = slab * SLABK;

      // ---- stage weight slab: fp32 [k][n] -> bf16 hi/lo transposed [n][k] ----
      {
        const int n4 = (tid & 15) * 4;   // 4 consecutive experts
        const int kr = tid >> 4;         // 16 k-rows per pass
        #pragma unroll
        for (int p = 0; p < SLABK / 16; ++p) {
          const int kl = kr + p * 16;
          float4 wv = *(const float4*)(w + (size_t)(kk + kl) * NE + n4);
          if (slab + 1 < NSLAB)          // warm L2->L0 for next slab (global_prefetch_b8)
            __builtin_prefetch(w + (size_t)(kk + SLABK + kl) * NE + n4, 0, 1);
          #pragma unroll
          for (int j = 0; j < 4; ++j) {
            const float f = (&wv.x)[j];
            const __bf16 h = (__bf16)f;                 // RNE
            const __bf16 l = (__bf16)(f - (float)h);    // residual
            sHi[(n4 + j) * PITCH + kl] = h;
            sLo[(n4 + j) * PITCH + kl] = l;
          }
        }
      }
      __syncthreads();

      if (active) {
        #pragma unroll
        for (int ch = 0; ch < SLABK / 32; ++ch) {
          const int k0 = kk + ch * 32;

          // ---- A fragment: 16 fp32 -> bf16 hi/lo (two 32B contiguous loads) ----
          const float* ap = aRow + k0 + koffA;
          float4 a0 = *(const float4*)(ap);
          float4 a1 = *(const float4*)(ap + 4);
          float4 a2 = *(const float4*)(ap + 16);
          float4 a3 = *(const float4*)(ap + 20);
          float av[16];
          av[0]=a0.x; av[1]=a0.y; av[2]=a0.z;  av[3]=a0.w;
          av[4]=a1.x; av[5]=a1.y; av[6]=a1.z;  av[7]=a1.w;
          av[8]=a2.x; av[9]=a2.y; av[10]=a2.z; av[11]=a2.w;
          av[12]=a3.x; av[13]=a3.y; av[14]=a3.z; av[15]=a3.w;
          v16bf ahi, alo;
          #pragma unroll
          for (int e = 0; e < 16; ++e) {
            const float  f = av[e];
            const __bf16 h = (__bf16)f;
            ahi[e] = h;
            alo[e] = (__bf16)(f - (float)h);
          }

          // ---- B fragments from LDS + 3 split-WMMAs per column tile ----
          const int kb = ch * 32 + hl * 8;   // B layout: lanes0-15 K{0..7,16..23}, lanes16-31 +8
          #pragma unroll
          for (int t = 0; t < 4; ++t) {
            const __bf16* ph = &sHi[(t * 16 + l16) * PITCH + kb];
            const __bf16* pl = &sLo[(t * 16 + l16) * PITCH + kb];
            v8bf bh0 = *(const v8bf*)(ph);
            v8bf bh1 = *(const v8bf*)(ph + 16);
            v8bf bl0 = *(const v8bf*)(pl);
            v8bf bl1 = *(const v8bf*)(pl + 16);
            v16bf bhi = __builtin_shufflevector(bh0, bh1, 0,1,2,3,4,5,6,7,8,9,10,11,12,13,14,15);
            v16bf blo = __builtin_shufflevector(bl0, bl1, 0,1,2,3,4,5,6,7,8,9,10,11,12,13,14,15);
            c[t] = __builtin_amdgcn_wmma_f32_16x16x32_bf16(false, ahi, false, bhi,
                                                           (short)0, c[t], false, false);
            c[t] = __builtin_amdgcn_wmma_f32_16x16x32_bf16(false, alo, false, bhi,
                                                           (short)0, c[t], false, false);
            c[t] = __builtin_amdgcn_wmma_f32_16x16x32_bf16(false, ahi, false, blo,
                                                           (short)0, c[t], false, false);
          }
        }
      }
      __syncthreads();
    }

    // ---- epilogue: top-8 + softmax. C layout: VGPR j = row j (lanes0-15) /
    // row j+8 (lanes16-31); column = t*16 + l16. 16-lane butterflies keep the
    // two halves' rows independent.
    if (active) {
      #pragma unroll
      for (int j = 0; j < 8; ++j) {
        float cv0 = c[0][j], cv1 = c[1][j], cv2 = c[2][j], cv3 = c[3][j];
        float tv[TOPK]; int ti[TOPK];
        #pragma unroll
        for (int r = 0; r < TOPK; ++r) {
          float bv = cv0; int bi = l16;                      // '>' keeps lowest idx on ties
          if (cv1 > bv) { bv = cv1; bi = l16 + 16; }
          if (cv2 > bv) { bv = cv2; bi = l16 + 32; }
          if (cv3 > bv) { bv = cv3; bi = l16 + 48; }
          #pragma unroll
          for (int m = 1; m <= 8; m <<= 1) {                 // stays inside 16-lane half
            const float ov = __shfl_xor(bv, m, 32);
            const int   oi = __shfl_xor(bi, m, 32);
            if (ov > bv || (ov == bv && oi < bi)) { bv = ov; bi = oi; }
          }
          tv[r] = bv; ti[r] = bi;
          if ((bi & 15) == l16) {                            // owner masks the winner out
            const int t = bi >> 4;
            const float ninf = -__builtin_inff();
            if      (t == 0) cv0 = ninf;
            else if (t == 1) cv1 = ninf;
            else if (t == 2) cv2 = ninf;
            else             cv3 = ninf;
          }
        }
        // softmax over the (sorted-desc) top-8
        const float mx = tv[0];
        float ev[TOPK]; float sum = 0.f;
        #pragma unroll
        for (int r = 0; r < TOPK; ++r) { ev[r] = __expf(tv[r] - mx); sum += ev[r]; }
        const float inv = 1.0f / sum;

        if (l16 == 0) {                                      // lanes 0 and 16 write 2 rows
          const size_t rowOut = (size_t)(tile * 16 + j + 8 * hl);
          float4 s0 = make_float4(ev[0]*inv, ev[1]*inv, ev[2]*inv, ev[3]*inv);
          float4 s1 = make_float4(ev[4]*inv, ev[5]*inv, ev[6]*inv, ev[7]*inv);
          int4   e0 = make_int4(ti[0], ti[1], ti[2], ti[3]);
          int4   e1 = make_int4(ti[4], ti[5], ti[6], ti[7]);
          *(float4*)(outS + rowOut * TOPK)     = s0;
          *(float4*)(outS + rowOut * TOPK + 4) = s1;
          *(int4*)  (outE + rowOut * TOPK)     = e0;
          *(int4*)  (outE + rowOut * TOPK + 4) = e1;
        }
      }
    }
  }
}

extern "C" void kernel_launch(void* const* d_in, const int* in_sizes, int n_in,
                              void* d_out, int out_size, void* d_ws, size_t ws_size,
                              hipStream_t stream) {
  const float* x = (const float*)d_in[0];
  const float* w = (const float*)d_in[1];
  const int tokens = in_sizes[0] / D;          // 65536
  float* outS = (float*)d_out;                 // scores [tokens, 8] fp32
  int*   outE = (int*)(outS + (size_t)tokens * TOPK);  // experts [tokens, 8] i32
  const int ntiles = tokens / 16;              // 16 tokens per wave-tile

  const int blocks = 256;                      // grid-stride covers all tiles
  moe_gate_wmma<<<blocks, THREADS, 0, stream>>>(x, w, outS, outE, ntiles);
}
